// ProbAttention_55705725829697
// MI455X (gfx1250) — compile-verified, matrix operand order checked by reading, not run on hardware
//
#include <hip/hip_runtime.h>
#include <hip/hip_bf16.h>

typedef __bf16 bf16;
typedef __attribute__((ext_vector_type(16))) __bf16 v16bf;
typedef __attribute__((ext_vector_type(8)))  float  v8f;

#define WMMA_BF16(A, B, C) \
  __builtin_amdgcn_wmma_f32_16x16x32_bf16(false, (A), false, (B), (short)0, (C), false, false)

static constexpr int BB = 8, LL = 4096, DM = 512, HH = 8, DK = 64, DV = 64, UU = 41;
static constexpr int NBH = BB * HH;               // 64
static constexpr int MROWS = BB * LL;             // 32768

// ---- workspace layout (bytes) ----
static constexpr size_t OFF_QB  = 0;                                     // Q bf16 head-major [b][h][l][64]
static constexpr size_t OFF_KB  = OFF_QB + (size_t)NBH * LL * DK * 2;    // K bf16
static constexpr size_t OFF_VB  = OFF_KB + (size_t)NBH * LL * DK * 2;    // V bf16
static constexpr size_t OFF_PRE = 0;                                     // pre-LN f32, overlays Q/K (dead by then)
static constexpr size_t OFF_M   = OFF_VB + (size_t)NBH * LL * DV * 2;    // sparsity metric f32 [bh][l]
static constexpr size_t OFF_IDX = OFF_M + (size_t)NBH * LL * 4;          // top-k idx int [bh][48]
static constexpr size_t OFF_VME = OFF_IDX + (size_t)NBH * 48 * 4;        // vmean f32 [bh][64]
static constexpr size_t OFF_CTX = OFF_VME + (size_t)NBH * DV * 4;        // context bf16 [b*l][512]
static constexpr size_t OFF_WT  = OFF_CTX + (size_t)MROWS * DM * 2;      // 4x transposed bf16 weights

// ---- WMMA 16-bit fragment load from LDS (A layout; B uses mirrored layout via
// N-major storage, so same loader works). stride must be a multiple of 8 elems.
__device__ __forceinline__ v16bf lds_frag(const bf16* base, int stride) {
  const int lane = threadIdx.x & 31;
  const int r = lane & 15, hf = lane >> 4;
  const bf16* p = base + r * stride + hf * 8;
  union { v16bf v; uint4 u[2]; } t;
  t.u[0] = *(const uint4*)(p);
  t.u[1] = *(const uint4*)(p + 16);
  return t.v;
}

__device__ __forceinline__ void zero8(v8f& a) {
#pragma unroll
  for (int e = 0; e < 8; ++e) a[e] = 0.0f;
}

// async copy of 32B global(bf16) -> LDS, tracked by ASYNCcnt
__device__ __forceinline__ void async_copy32(bf16* lds_dst, const bf16* gsrc) {
  unsigned loff = (unsigned)(uintptr_t)lds_dst;   // LDS aperture keeps offset in addr[31:0]
  asm volatile(
      "global_load_async_to_lds_b128 %0, %1, off\n\t"
      "global_load_async_to_lds_b128 %0, %1, off offset:16"
      :
      : "v"(loff), "v"(gsrc)
      : "memory");
}

// async copy of 64B global(bf16) -> LDS
__device__ __forceinline__ void async_copy128(bf16* lds_dst, const bf16* gsrc) {
  unsigned loff = (unsigned)(uintptr_t)lds_dst;
  asm volatile(
      "global_load_async_to_lds_b128 %0, %1, off\n\t"
      "global_load_async_to_lds_b128 %0, %1, off offset:16\n\t"
      "global_load_async_to_lds_b128 %0, %1, off offset:32\n\t"
      "global_load_async_to_lds_b128 %0, %1, off offset:48"
      :
      : "v"(loff), "v"(gsrc)
      : "memory");
}

// ============ weight convert: Wt[n][k] = bf16(W[k][n]) ============
__global__ __launch_bounds__(256) void k_convw(const float* __restrict__ W,
                                               bf16* __restrict__ Wt) {
  int t = blockIdx.x * 256 + threadIdx.x;       // 512*512 elements
  int n = t >> 9, k = t & 511;
  Wt[t] = (bf16)W[k * 512 + n];
}

// ============ projection GEMM: (MROWSx512 f32) @ Wt -> bf16 head-major ============
__global__ __launch_bounds__(256) void k_proj_gemm(const float* __restrict__ A,
                                                   const bf16* __restrict__ Bt,
                                                   bf16* __restrict__ Dst) {
  __shared__ bf16 As[128 * 40];
  __shared__ bf16 Bs[128 * 40];
  const int tid = threadIdx.x;
  const int wave = tid >> 5, lane = tid & 31;
  const int wm = wave >> 1, wn = wave & 1;      // 4x2 wave grid
  const int rowBase = blockIdx.y * 128;
  const int colBase = blockIdx.x * 128;
  const int ldRow = tid >> 1, ldHalf = tid & 1;

  v8f acc[2][4];
#pragma unroll
  for (int i = 0; i < 2; ++i)
#pragma unroll
    for (int j = 0; j < 4; ++j) zero8(acc[i][j]);

  for (int k0 = 0; k0 < 512; k0 += 32) {
    __syncthreads();
    // A tile: 16 f32 per thread -> cvt -> bf16 LDS
    const float* ga = A + (size_t)(rowBase + ldRow) * 512 + k0 + ldHalf * 16;
    if (k0 + 32 < 512) __builtin_prefetch(ga + 32, 0, 0);
    bf16* sa = As + ldRow * 40 + ldHalf * 16;
#pragma unroll
    for (int j = 0; j < 16; j += 4) {
      float4 f = *(const float4*)(ga + j);
      sa[j + 0] = (bf16)f.x; sa[j + 1] = (bf16)f.y;
      sa[j + 2] = (bf16)f.z; sa[j + 3] = (bf16)f.w;
    }
    // B tile: async DMA straight into LDS
    const bf16* gb = Bt + (size_t)(colBase + ldRow) * 512 + k0 + ldHalf * 16;
    async_copy32(Bs + ldRow * 40 + ldHalf * 16, gb);
    asm volatile("s_wait_asynccnt 0x0" ::: "memory");
    __syncthreads();

    v16bf af[2], bf[4];
#pragma unroll
    for (int i = 0; i < 2; ++i) af[i] = lds_frag(As + (wm * 32 + i * 16) * 40, 40);
#pragma unroll
    for (int j = 0; j < 4; ++j) bf[j] = lds_frag(Bs + (wn * 64 + j * 16) * 40, 40);
#pragma unroll
    for (int i = 0; i < 2; ++i)
#pragma unroll
      for (int j = 0; j < 4; ++j) acc[i][j] = WMMA_BF16(af[i], bf[j], acc[i][j]);
  }

  const int ln = lane & 15, lh = lane >> 4;
#pragma unroll
  for (int i = 0; i < 2; ++i)
#pragma unroll
    for (int j = 0; j < 4; ++j) {
      int col = colBase + wn * 64 + j * 16 + ln;
      int h = col >> 6, d = col & 63;
#pragma unroll
      for (int v = 0; v < 8; ++v) {
        int row = rowBase + wm * 32 + i * 16 + lh * 8 + v;
        int b = row >> 12, l = row & 4095;
        Dst[(((size_t)(b * HH + h)) * LL + l) * 64 + d] = (bf16)acc[i][j][v];
      }
    }
}

// ============ FC GEMM: (ctx bf16) @ WfcT + residual -> f32 pre-LN ============
// Double-buffered async-DMA pipeline: tile t+1 streams into LDS while WMMAs run on tile t.
__global__ __launch_bounds__(256) void k_fc_gemm(const bf16* __restrict__ A,
                                                 const bf16* __restrict__ Bt,
                                                 const float* __restrict__ resid,
                                                 float* __restrict__ pre) {
  __shared__ bf16 As[2][128 * 40];
  __shared__ bf16 Bs[2][128 * 40];
  const int tid = threadIdx.x;
  const int wave = tid >> 5, lane = tid & 31;
  const int wm = wave >> 1, wn = wave & 1;
  const int rowBase = blockIdx.y * 128;
  const int colBase = blockIdx.x * 128;
  const int ldRow = tid >> 1, ldHalf = tid & 1;

  v8f acc[2][4];
#pragma unroll
  for (int i = 0; i < 2; ++i)
#pragma unroll
    for (int j = 0; j < 4; ++j) zero8(acc[i][j]);

  auto issue = [&](int t, int buf) {
    const bf16* ga = A + (size_t)(rowBase + ldRow) * 512 + t * 32 + ldHalf * 16;
    const bf16* gb = Bt + (size_t)(colBase + ldRow) * 512 + t * 32 + ldHalf * 16;
    async_copy32(&As[buf][ldRow * 40 + ldHalf * 16], ga);
    async_copy32(&Bs[buf][ldRow * 40 + ldHalf * 16], gb);
  };

  issue(0, 0);
  for (int t = 0; t < 16; ++t) {
    const int buf = t & 1;
    if (t + 1 < 16) {
      issue(t + 1, buf ^ 1);                      // overlap next tile's DMA
      asm volatile("s_wait_asynccnt 0x4" ::: "memory");  // tile t's 4 ops retired
    } else {
      asm volatile("s_wait_asynccnt 0x0" ::: "memory");
    }
    __syncthreads();

    v16bf af[2], bf[4];
#pragma unroll
    for (int i = 0; i < 2; ++i) af[i] = lds_frag(&As[buf][(wm * 32 + i * 16) * 40], 40);
#pragma unroll
    for (int j = 0; j < 4; ++j) bf[j] = lds_frag(&Bs[buf][(wn * 64 + j * 16) * 40], 40);
#pragma unroll
    for (int i = 0; i < 2; ++i)
#pragma unroll
      for (int j = 0; j < 4; ++j) acc[i][j] = WMMA_BF16(af[i], bf[j], acc[i][j]);
    __syncthreads();                              // all reads done before buf is re-targeted
  }

  const int ln = lane & 15, lh = lane >> 4;
#pragma unroll
  for (int i = 0; i < 2; ++i)
#pragma unroll
    for (int j = 0; j < 4; ++j) {
      int col = colBase + wn * 64 + j * 16 + ln;
#pragma unroll
      for (int v = 0; v < 8; ++v) {
        int row = rowBase + wm * 32 + i * 16 + lh * 8 + v;
        pre[(size_t)row * 512 + col] = acc[i][j][v] + resid[(size_t)row * 512 + col];
      }
    }
}

// ============ sampled QK metric: M = max - mean over 41 sampled keys ============
__global__ __launch_bounds__(256) void k_sample(const bf16* __restrict__ Qb,
                                                const bf16* __restrict__ Kb,
                                                const int* __restrict__ samp,
                                                float* __restrict__ Mout) {
  int t = blockIdx.x * 256 + threadIdx.x;      // NBH*LL
  int l = t & 4095, bh = t >> 12;
  const bf16* q = Qb + ((size_t)bh * LL + l) * 64;
  float qf[64];
#pragma unroll
  for (int d = 0; d < 64; ++d) qf[d] = (float)q[d];
  float mx = -1e30f, sm = 0.f;
  for (int s = 0; s < UU; ++s) {
    int kl = samp[l * UU + s];
    const bf16* kr = Kb + ((size_t)bh * LL + kl) * 64;
    float a = 0.f;
#pragma unroll
    for (int d = 0; d < 64; ++d) a += qf[d] * (float)kr[d];
    mx = fmaxf(mx, a);
    sm += a;
  }
  Mout[t] = mx - sm * (1.0f / UU);
}

// ============ top-41 per (b,h) via iterative argmax ============
__global__ __launch_bounds__(256) void k_topk(float* __restrict__ Mv,
                                              int* __restrict__ idxOut) {
  __shared__ float sv[256];
  __shared__ int   si[256];
  const int bh = blockIdx.x, tid = threadIdx.x;
  float* Mrow = Mv + (size_t)bh * LL;
  for (int u = 0; u < UU; ++u) {
    float best = -1e30f; int bi = 0;
    for (int j = tid; j < LL; j += 256) {
      float v = Mrow[j];
      if (v > best) { best = v; bi = j; }
    }
    sv[tid] = best; si[tid] = bi;
    __syncthreads();
    for (int s = 128; s > 0; s >>= 1) {
      if (tid < s && sv[tid + s] > sv[tid]) { sv[tid] = sv[tid + s]; si[tid] = si[tid + s]; }
      __syncthreads();
    }
    if (tid == 0) { idxOut[bh * 48 + u] = si[0]; Mrow[si[0]] = -1e30f; }
    __syncthreads();
  }
}

// ============ V mean per (b,h,d) ============
__global__ __launch_bounds__(256) void k_vmean(const bf16* __restrict__ Vb,
                                               float* __restrict__ vmean) {
  __shared__ float red[256];
  const int bh = blockIdx.x, tid = threadIdx.x;
  const int d = tid & 63, part = tid >> 6;     // 4 parts of 1024 rows
  const bf16* v = Vb + (size_t)bh * LL * 64;
  float s = 0.f;
  for (int l = part * 1024; l < (part + 1) * 1024; ++l) s += (float)v[(size_t)l * 64 + d];
  red[tid] = s;
  __syncthreads();
  if (part == 0)
    vmean[bh * 64 + d] = (red[d] + red[64 + d] + red[128 + d] + red[192 + d]) * (1.0f / LL);
}

// ============ context fill with broadcast V-mean (bf16) ============
__global__ __launch_bounds__(256) void k_fillctx(const float* __restrict__ vmean,
                                                 bf16* __restrict__ ctx) {
  size_t t = (size_t)blockIdx.x * 256 + threadIdx.x;   // MROWS*512 elements
  int c = (int)(t & 511);
  size_t bl = t >> 9;
  int b = (int)(bl >> 12);
  int h = c >> 6, d = c & 63;
  ctx[t] = (bf16)vmean[(b * HH + h) * 64 + d];
}

// ============ sparse attention (flash over 128-key chunks), scatter into ctx ============
__global__ __launch_bounds__(128) void k_attention(const bf16* __restrict__ Qb,
                                                   const bf16* __restrict__ Kb,
                                                   const bf16* __restrict__ Vb,
                                                   const int* __restrict__ idxTop,
                                                   bf16* __restrict__ ctx) {
  __shared__ bf16  Qs[48 * 72];
  __shared__ bf16  Ks[128 * 72];
  __shared__ float Sl[48 * 132];
  __shared__ bf16  P[48 * 136];
  __shared__ bf16  Vt[64 * 136];
  __shared__ float rowM[48], rowL[48], rowScale[48];
  __shared__ int   sIdx[48];

  const int bh = blockIdx.x;                 // b*8+h
  const int tid = threadIdx.x;               // 128 threads = 4 waves
  const int wave = tid >> 5, lane = tid & 31;
  const int ln = lane & 15, lh = lane >> 4;
  const bf16* Qh = Qb + (size_t)bh * LL * 64;
  const bf16* Kh = Kb + (size_t)bh * LL * 64;
  const bf16* Vh = Vb + (size_t)bh * LL * 64;

  if (tid < 48) {
    int valid = (tid < UU);
    int l = valid ? idxTop[bh * 48 + tid] : 0;
    sIdx[tid] = l;
    const bf16* src = Qh + (size_t)l * 64;
    bf16* dst = Qs + tid * 72;
    if (valid) {
#pragma unroll
      for (int j = 0; j < 64; j += 8) *(uint4*)(dst + j) = *(const uint4*)(src + j);
    } else {
      uint4 z = make_uint4(0, 0, 0, 0);
#pragma unroll
      for (int j = 0; j < 64; j += 8) *(uint4*)(dst + j) = z;
    }
    rowM[tid] = -1e30f;
    rowL[tid] = 0.f;
  }

  v8f accO[3];
#pragma unroll
  for (int i = 0; i < 3; ++i) zero8(accO[i]);

  for (int kc = 0; kc < LL; kc += 128) {
    __syncthreads();
    // K tile via async DMA (natural layout = B operand); V transposed via VALU
    async_copy128(Ks + tid * 72, Kh + (size_t)(kc + tid) * 64);
    {
      const bf16* vs = Vh + (size_t)(kc + tid) * 64;
#pragma unroll
      for (int d = 0; d < 64; ++d) Vt[d * 136 + tid] = vs[d];
    }
    asm volatile("s_wait_asynccnt 0x0" ::: "memory");
    __syncthreads();

    // scores: each wave -> 2 key tiles x 3 query tiles, K-loop over DK (2x32)
#pragma unroll
    for (int jn = 0; jn < 2; ++jn) {
      const int nt = wave * 2 + jn;
      v16bf b0 = lds_frag(Ks + nt * 16 * 72 + 0, 72);
      v16bf b1 = lds_frag(Ks + nt * 16 * 72 + 32, 72);
#pragma unroll
      for (int mt = 0; mt < 3; ++mt) {
        v8f s; zero8(s);
        v16bf a0 = lds_frag(Qs + mt * 16 * 72 + 0, 72);
        s = WMMA_BF16(a0, b0, s);
        v16bf a1 = lds_frag(Qs + mt * 16 * 72 + 32, 72);
        s = WMMA_BF16(a1, b1, s);
        const int n = nt * 16 + ln;
        const int mb = mt * 16 + lh * 8;
#pragma unroll
        for (int v = 0; v < 8; ++v) Sl[(mb + v) * 132 + n] = s[v] * 0.125f;  // 1/sqrt(64)
      }
    }
    __syncthreads();

    // online softmax per row
    if (tid < 48) {
      float mold = rowM[tid], mloc = mold;
      const float* srow = Sl + tid * 132;
      for (int j = 0; j < 128; ++j) mloc = fmaxf(mloc, srow[j]);
      float scale = __expf(mold - mloc);
      float psum = 0.f;
      bf16* prow = P + tid * 136;
      for (int j = 0; j < 128; ++j) {
        float p = __expf(srow[j] - mloc);
        psum += p;
        prow[j] = (bf16)p;
      }
      rowM[tid] = mloc;
      rowL[tid] = rowL[tid] * scale + psum;
      rowScale[tid] = scale;
    }
    __syncthreads();

    // rescale running output, then P @ V (wave owns one 16-wide dv tile)
#pragma unroll
    for (int mt = 0; mt < 3; ++mt) {
      const int mb = mt * 16 + lh * 8;
#pragma unroll
      for (int v = 0; v < 8; ++v) accO[mt][v] *= rowScale[mb + v];
    }
#pragma unroll
    for (int kk = 0; kk < 128; kk += 32) {
      v16bf bf = lds_frag(Vt + (wave * 16) * 136 + kk, 136);
#pragma unroll
      for (int mt = 0; mt < 3; ++mt) {
        v16bf af = lds_frag(P + mt * 16 * 136 + kk, 136);
        accO[mt] = WMMA_BF16(af, bf, accO[mt]);
      }
    }
  }
  __syncthreads();

  // scatter normalized rows u<41 into context
  const int b = bh >> 3, h = bh & 7;
  const int dv = wave * 16 + ln;
#pragma unroll
  for (int mt = 0; mt < 3; ++mt) {
#pragma unroll
    for (int v = 0; v < 8; ++v) {
      const int u = mt * 16 + lh * 8 + v;
      if (u < UU) {
        const int l = sIdx[u];
        float o = accO[mt][v] / rowL[u];
        ctx[((size_t)(b * LL + l)) * 512 + h * 64 + dv] = (bf16)o;
      }
    }
  }
}

// ============ LayerNorm over 512 cols ============
__global__ __launch_bounds__(256) void k_ln(const float* __restrict__ pre,
                                            const float* __restrict__ gamma,
                                            const float* __restrict__ beta,
                                            float* __restrict__ out) {
  __shared__ float s1[256], s2[256];
  const size_t row = blockIdx.x;
  const int tid = threadIdx.x;
  const float* x = pre + row * 512;
  float a = x[tid], b = x[tid + 256];
  s1[tid] = a + b;
  s2[tid] = a * a + b * b;
  __syncthreads();
  for (int s = 128; s > 0; s >>= 1) {
    if (tid < s) { s1[tid] += s1[tid + s]; s2[tid] += s2[tid + s]; }
    __syncthreads();
  }
  const float mu = s1[0] * (1.0f / 512.0f);
  const float var = s2[0] * (1.0f / 512.0f) - mu * mu;
  const float inv = rsqrtf(var + 1e-5f);
  out[row * 512 + tid]       = gamma[tid] * ((a - mu) * inv) + beta[tid];
  out[row * 512 + tid + 256] = gamma[tid + 256] * ((b - mu) * inv) + beta[tid + 256];
}

extern "C" void kernel_launch(void* const* d_in, const int* in_sizes, int n_in,
                              void* d_out, int out_size, void* d_ws, size_t ws_size,
                              hipStream_t stream) {
  const float* inQ   = (const float*)d_in[0];
  const float* inK   = (const float*)d_in[1];
  const float* inV   = (const float*)d_in[2];
  const float* WQ    = (const float*)d_in[3];
  const float* WK    = (const float*)d_in[4];
  const float* WV    = (const float*)d_in[5];
  const float* Wfc   = (const float*)d_in[6];
  const float* gamma = (const float*)d_in[7];
  const float* beta  = (const float*)d_in[8];
  const int*   samp  = (const int*)d_in[9];

  char* ws = (char*)d_ws;
  bf16*  Qb    = (bf16*)(ws + OFF_QB);
  bf16*  Kb    = (bf16*)(ws + OFF_KB);
  bf16*  Vb    = (bf16*)(ws + OFF_VB);
  float* preW  = (float*)(ws + OFF_PRE);
  float* Mw    = (float*)(ws + OFF_M);
  int*   idxW  = (int*)(ws + OFF_IDX);
  float* vmeW  = (float*)(ws + OFF_VME);
  bf16*  ctx   = (bf16*)(ws + OFF_CTX);
  bf16*  wtQ   = (bf16*)(ws + OFF_WT);
  bf16*  wtK   = wtQ + 512 * 512;
  bf16*  wtV   = wtK + 512 * 512;
  bf16*  wtFC  = wtV + 512 * 512;

  // weights -> bf16, transposed to N-major
  k_convw<<<(512 * 512) / 256, 256, 0, stream>>>(WQ, wtQ);
  k_convw<<<(512 * 512) / 256, 256, 0, stream>>>(WK, wtK);
  k_convw<<<(512 * 512) / 256, 256, 0, stream>>>(WV, wtV);
  k_convw<<<(512 * 512) / 256, 256, 0, stream>>>(Wfc, wtFC);

  // projections
  dim3 gGemm(DM / 128, MROWS / 128);
  k_proj_gemm<<<gGemm, 256, 0, stream>>>(inQ, wtQ, Qb);
  k_proj_gemm<<<gGemm, 256, 0, stream>>>(inK, wtK, Kb);
  k_proj_gemm<<<gGemm, 256, 0, stream>>>(inV, wtV, Vb);

  // sparsity metric + top-k + V mean + context fill
  k_sample<<<(NBH * LL) / 256, 256, 0, stream>>>(Qb, Kb, samp, Mw);
  k_topk<<<NBH, 256, 0, stream>>>(Mw, idxW);
  k_vmean<<<NBH, 256, 0, stream>>>(Vb, vmeW);
  k_fillctx<<<(int)(((size_t)MROWS * 512) / 256), 256, 0, stream>>>(vmeW, ctx);

  // sparse attention, scatter into context
  k_attention<<<NBH, 128, 0, stream>>>(Qb, Kb, Vb, idxW, ctx);

  // output projection + residual, then LayerNorm
  k_fc_gemm<<<gGemm, 256, 0, stream>>>(ctx, wtFC, inQ, preW);
  k_ln<<<MROWS, 256, 0, stream>>>(preW, gamma, beta, (float*)d_out);
}
